// AttnLSTMDecoder_25709674234379
// MI455X (gfx1250) — compile-verified, matrix operand order checked by reading, not run on hardware
//
#include <hip/hip_runtime.h>
#include <hip/hip_bf16.h>

#define VOCAB 128
#define STATE 100
#define ATTN  100
#define EMB   100
#define LLEN  4096
#define TSTEPS 1024

#define NB   64          // blocks in decode kernel (must all be co-resident)
#define TPB  256         // threads per block (8 wave32)
#define RPB  (LLEN / NB) // 64 rows per block

#define KPAD 224         // 200 padded to 7*32
#define NPAD 112         // 100 padded to 7*16

// ---- workspace layout (float offsets) ----
#define WS_P    0                    // P = tanh-arg precompute [4096*100]
#define WS_Q    409600               // q[100]  (W1_h @ [h;c])
#define WS_H    409700               // h[100]
#define WS_C    409800               // c[100]
#define WS_PE   409900               // prev_emb[100]
#define WS_MB   410000               // per-block softmax max [NB]
#define WS_ZB   410064               // per-block sum-exp [NB]
#define WS_CTX  410128               // per-block partial context [NB][200]
#define WS_CNT  422944               // grid-barrier counter (unsigned)
#define WS_IMH  422948               // f16 im, padded [4096][224]  (458752 floats)
#define WS_W1HT 881700               // f16 W1[:, :200]^T-ish, [112][224] (12544 floats)
// total ~894244 floats ~ 3.6 MB

typedef __attribute__((ext_vector_type(16))) _Float16 v16h;
typedef __attribute__((ext_vector_type(8)))  _Float16 v8h;
typedef __attribute__((ext_vector_type(8)))  float    v8f;

// ---------------------------------------------------------------------------
// Init: zero barrier counter + state; prev_emb = emb[EOS_ID=0]; q = 0 (h=c=0)
// ---------------------------------------------------------------------------
__global__ void init_kernel(float* __restrict__ ws, const float* __restrict__ emb) {
    int tid = threadIdx.x;
    if (tid < 100) {
        ws[WS_Q  + tid] = 0.f;
        ws[WS_H  + tid] = 0.f;
        ws[WS_C  + tid] = 0.f;
        ws[WS_PE + tid] = emb[tid];      // emb row 0 (EOS)
    }
    if (tid == 0) *((unsigned*)(ws + WS_CNT)) = 0u;
}

// ---------------------------------------------------------------------------
// Pack: f32 -> f16 with zero padding so the WMMA kernel does pure vector loads.
//   imh [4096][224]  : imh[l][k]   = k<200 ? im[l,k] : 0
//   w1ht[112][224]   : w1ht[n][k]  = (n<100 && k<200) ? W1[n,k] : 0
// ---------------------------------------------------------------------------
__global__ void pack_kernel(const float* __restrict__ im, const float* __restrict__ W1,
                            _Float16* __restrict__ imh, _Float16* __restrict__ w1ht) {
    const int stride = gridDim.x * blockDim.x;
    const int idx = blockIdx.x * blockDim.x + threadIdx.x;
    for (int i = idx; i < LLEN * KPAD; i += stride) {
        const int l = i / KPAD, k = i % KPAD;
        imh[i] = (k < 200) ? (_Float16)im[l * 200 + k] : (_Float16)0.f;
    }
    for (int i = idx; i < NPAD * KPAD; i += stride) {
        const int n = i / KPAD, k = i % KPAD;
        w1ht[i] = (n < ATTN && k < 200) ? (_Float16)W1[n * 400 + k] : (_Float16)0.f;
    }
}

// ---------------------------------------------------------------------------
// Precompute P[l,n] = sum_{k<200} im[l,k]*W1[n,k] + b1[n]
// One wave per 16-row M tile; computes all 7 N tiles (A frag reused 7x).
// A fragment for lane (half=lane/16, lr=lane%16): elements 0..7 = K run at
// kb+8*half, elements 8..15 = K run at kb+16+8*half  -> two 16B loads.
// B fragment identical with row lr of w1ht tile n.
// ---------------------------------------------------------------------------
__global__ __launch_bounds__(32) void precompute_P(const _Float16* __restrict__ imh,
                                                   const _Float16* __restrict__ w1ht,
                                                   const float* __restrict__ b1,
                                                   float* __restrict__ P) {
    const int lane = threadIdx.x;
    const int half = lane >> 4;      // 0|1
    const int lr   = lane & 15;      // 0..15
    const int m0   = blockIdx.x * 16;

    v8f acc[7];
#pragma unroll
    for (int n = 0; n < 7; ++n) acc[n] = (v8f){};

    const v8h* arow = (const v8h*)(imh + (m0 + lr) * KPAD);

#pragma unroll
    for (int kc = 0; kc < 7; ++kc) {
        const int o0 = (kc * 32 + 8 * half) >> 3;    // v8h index
        const v8h alo = arow[o0];
        const v8h ahi = arow[o0 + 2];
        const v16h a = __builtin_shufflevector(alo, ahi,
                        0,1,2,3,4,5,6,7,8,9,10,11,12,13,14,15);
#pragma unroll
        for (int n = 0; n < 7; ++n) {
            const v8h* brow = (const v8h*)(w1ht + (n * 16 + lr) * KPAD);
            const v8h blo = brow[o0];
            const v8h bhi = brow[o0 + 2];
            const v16h b = __builtin_shufflevector(blo, bhi,
                            0,1,2,3,4,5,6,7,8,9,10,11,12,13,14,15);
            acc[n] = __builtin_amdgcn_wmma_f32_16x16x32_f16(false, a, false, b,
                                                            (short)0, acc[n],
                                                            false, false);
        }
    }

#pragma unroll
    for (int n = 0; n < 7; ++n) {
        const int col = n * 16 + lr;                 // C/D: N = lane%16
        if (col < ATTN) {
            const float bb = b1[col];
#pragma unroll
            for (int i = 0; i < 8; ++i)              // C/D: M = i + 8*(lane/16)
                P[(m0 + i + 8 * half) * ATTN + col] = acc[n][i] + bb;
        }
    }
}

// ---------------------------------------------------------------------------
// Monotonic agent-scope grid barrier (all NB blocks co-resident).
// ---------------------------------------------------------------------------
__device__ __forceinline__ void grid_barrier(unsigned* cnt, unsigned target) {
    __syncthreads();
    if (threadIdx.x == 0) {
        __hip_atomic_fetch_add(cnt, 1u, __ATOMIC_ACQ_REL, __HIP_MEMORY_SCOPE_AGENT);
        while (__hip_atomic_load(cnt, __ATOMIC_ACQUIRE, __HIP_MEMORY_SCOPE_AGENT) < target)
            __builtin_amdgcn_s_sleep(2);
    }
    __syncthreads();
}

// ---------------------------------------------------------------------------
// Persistent decode kernel: all T=1024 steps in one launch.
// ---------------------------------------------------------------------------
__global__ __launch_bounds__(TPB) void decode_kernel(
    const float* __restrict__ im,   const int*   __restrict__ out_ids,
    const float* __restrict__ emb,  const float* __restrict__ W1,
    const float* __restrict__ W2,   const float* __restrict__ b2,
    const float* __restrict__ W_ih, const float* __restrict__ W_hh,
    const float* __restrict__ b_ih, const float* __restrict__ b_hh,
    const float* __restrict__ W_out,const float* __restrict__ b_out,
    float* __restrict__ ws, float* __restrict__ out)
{
    __shared__ float qs[ATTN];
    __shared__ float srow[RPB];
    __shared__ float wrow[RPB];
    __shared__ float sM[2];
    __shared__ float vec[300];       // [context(200) | prev_emb(100)]
    __shared__ float gates[400];
    __shared__ float h_s[STATE], c_s[STATE], h2_s[STATE], c2_s[STATE];
    __shared__ float logit_s[VOCAB];

    float* P    = ws + WS_P;
    float* q    = ws + WS_Q;
    float* hg   = ws + WS_H;
    float* cg   = ws + WS_C;
    float* pe   = ws + WS_PE;
    float* mbA  = ws + WS_MB;
    float* zbA  = ws + WS_ZB;
    float* ctxp = ws + WS_CTX;
    unsigned* cnt = (unsigned*)(ws + WS_CNT);

    const int tid  = threadIdx.x;
    const int blk  = blockIdx.x;
    const int base = blk * RPB;
    const int wave = tid >> 5;
    const int lane = tid & 31;

    float nll_sum = 0.f;

    for (int t = 0; t < TSTEPS; ++t) {
        // ---------------- Phase A: per-block attention partials --------------
        if (tid < ATTN) qs[tid] = q[tid];
        __syncthreads();

        // 8 waves x 8 rows: scores s_l = b2 + sum_n tanh(P[l,n]+q[n])*W2[n]
#pragma unroll 2
        for (int rr = 0; rr < RPB / (TPB / 32); ++rr) {
            const int lrow = wave * (RPB / (TPB / 32)) + rr;
            const float* Pr = P + (base + lrow) * ATTN;
            __builtin_prefetch(Pr + ATTN, 0, 1);           // global_prefetch next row
            float partial = 0.f;
            for (int n = lane; n < ATTN; n += 32)
                partial += tanhf(Pr[n] + qs[n]) * W2[n];
            for (int off = 16; off > 0; off >>= 1)
                partial += __shfl_down(partial, off, 32);
            if (lane == 0) srow[lrow] = partial + b2[0];
        }
        __syncthreads();

        if (tid == 0) {
            float mb = srow[0];
            for (int i = 1; i < RPB; ++i) mb = fmaxf(mb, srow[i]);
            sM[0] = mb;
        }
        __syncthreads();
        if (tid < RPB) wrow[tid] = __expf(srow[tid] - sM[0]);
        __syncthreads();
        if (tid == 0) {
            float z = 0.f;
            for (int i = 0; i < RPB; ++i) z += wrow[i];
            mbA[blk] = sM[0];
            zbA[blk] = z;
        }
        if (tid < 200) {                       // partial context (column tid)
            float acc = 0.f;
            const float* imc = im + base * 200 + tid;
            for (int i = 0; i < RPB; ++i) acc += wrow[i] * imc[i * 200];
            ctxp[blk * 200 + tid] = acc;
        }
        __threadfence();
        grid_barrier(cnt, (unsigned)NB * (2u * t + 1u));

        // ---------------- Phase B: block 0 serial tail ----------------------
        if (blk == 0) {
            if (tid == 0) {
                float M = mbA[0];
                for (int b = 1; b < NB; ++b) M = fmaxf(M, mbA[b]);
                sM[0] = M;
            }
            if (tid < STATE)  h_s[tid] = hg[tid];
            if (tid >= STATE && tid < 2 * STATE) c_s[tid - STATE] = cg[tid - STATE];
            __syncthreads();
            const float M = sM[0];
            if (tid < 200) {
                float acc = 0.f;
                for (int b = 0; b < NB; ++b)
                    acc += ctxp[b * 200 + tid] * __expf(mbA[b] - M);
                vec[tid] = acc;
            }
            if (tid == 0) {
                float Z = 0.f;
                for (int b = 0; b < NB; ++b) Z += zbA[b] * __expf(mbA[b] - M);
                sM[1] = Z;
            }
            __syncthreads();
            const float invZ = 1.f / sM[1];
            if (tid < 200) vec[tid] *= invZ;            // context
            if (tid < EMB) vec[200 + tid] = pe[tid];    // prev embedding
            __syncthreads();

            // gates = W_ih@vec + b_ih + W_hh@h + b_hh
            for (int g = tid; g < 4 * STATE; g += TPB) {
                float acc = b_ih[g] + b_hh[g];
                const float* wi = W_ih + g * 300;
                for (int k = 0; k < 300; ++k) acc += wi[k] * vec[k];
                const float* wh = W_hh + g * STATE;
                for (int k = 0; k < STATE; ++k) acc += wh[k] * h_s[k];
                gates[g] = acc;
            }
            __syncthreads();

            if (tid < STATE) {       // LSTM cell (i,f,g,o order)
                float ig = 1.f / (1.f + __expf(-gates[tid]));
                float fg = 1.f / (1.f + __expf(-gates[STATE + tid]));
                float gg = tanhf(gates[2 * STATE + tid]);
                float og = 1.f / (1.f + __expf(-gates[3 * STATE + tid]));
                float c2 = fg * c_s[tid] + ig * gg;
                float h2 = og * tanhf(c2);
                c2_s[tid] = c2;  h2_s[tid] = h2;
                cg[tid] = c2;    hg[tid] = h2;
            }
            __syncthreads();

            if (tid < VOCAB) {       // logits
                float acc = b_out[tid];
                const float* wo = W_out + tid * STATE;
                for (int k = 0; k < STATE; ++k) acc += wo[k] * h2_s[k];
                logit_s[tid] = acc;
            }
            // next-step q = W1[:,200:] @ [h2; c2]
            if (tid < ATTN) {
                float acc = 0.f;
                const float* w1r = W1 + tid * 400 + 200;
                for (int k = 0; k < STATE; ++k) acc += w1r[k] * h2_s[k];
                for (int k = 0; k < STATE; ++k) acc += w1r[STATE + k] * c2_s[k];
                q[tid] = acc;
            }
            __syncthreads();

            const int ch = out_ids[t];
            if (tid == 0) {
                float ml = logit_s[0];
                for (int v = 1; v < VOCAB; ++v) ml = fmaxf(ml, logit_s[v]);
                float z = 0.f;
                for (int v = 0; v < VOCAB; ++v) z += __expf(logit_s[v] - ml);
                nll_sum += __logf(z) + ml - logit_s[ch];
            }
            if (tid < EMB) pe[tid] = emb[ch * EMB + tid];
            __threadfence();
        }
        grid_barrier(cnt, (unsigned)NB * (2u * t + 2u));
    }

    if (blk == 0 && tid == 0) out[0] = nll_sum;
}

// ---------------------------------------------------------------------------
extern "C" void kernel_launch(void* const* d_in, const int* in_sizes, int n_in,
                              void* d_out, int out_size, void* d_ws, size_t ws_size,
                              hipStream_t stream) {
    const float* im    = (const float*)d_in[0];   // [1,4096,200]
    const int*   ids   = (const int*)  d_in[1];   // [1024]
    const float* emb   = (const float*)d_in[2];   // [128,100]
    const float* W1    = (const float*)d_in[3];   // [100,400]
    const float* b1    = (const float*)d_in[4];   // [100]
    const float* W2    = (const float*)d_in[5];   // [1,100]
    const float* b2    = (const float*)d_in[6];   // [1]
    const float* W_ih  = (const float*)d_in[7];   // [400,300]
    const float* W_hh  = (const float*)d_in[8];   // [400,100]
    const float* b_ih  = (const float*)d_in[9];   // [400]
    const float* b_hh  = (const float*)d_in[10];  // [400]
    const float* W_out = (const float*)d_in[11];  // [128,100]
    const float* b_out = (const float*)d_in[12];  // [128]
    float* ws  = (float*)d_ws;
    float* out = (float*)d_out;

    _Float16* imh  = (_Float16*)(ws + WS_IMH);
    _Float16* w1ht = (_Float16*)(ws + WS_W1HT);

    init_kernel<<<1, 256, 0, stream>>>(ws, emb);
    pack_kernel<<<512, 256, 0, stream>>>(im, W1, imh, w1ht);
    precompute_P<<<LLEN / 16, 32, 0, stream>>>(imh, w1ht, b1, ws + WS_P);
    decode_kernel<<<NB, TPB, 0, stream>>>(im, ids, emb, W1, W2, b2,
                                          W_ih, W_hh, b_ih, b_hh, W_out, b_out,
                                          ws, out);
}
